// Model_23132693856272
// MI455X (gfx1250) — compile-verified, matrix operand order checked by reading, not run on hardware
//
#include <hip/hip_runtime.h>
#include <hip/hip_bf16.h>
#include <stdint.h>

#define N_NODES   4096
#define H         500
#define HP        512
#define HEADS     5
#define DDEG      8
#define GATE_COLS 2000
#define GATE_PAD  2048
#define Z_PAD     2560     // 5 heads * 512 padded
#define KBLOCKS   16       // 512 / 32
#define SLOPE     0.2f

typedef __bf16 v16bf __attribute__((ext_vector_type(16)));
typedef float  v8f   __attribute__((ext_vector_type(8)));

union AFrag { v16bf v; uint32_t u[8]; };

static __device__ __forceinline__ unsigned short f2bf(float x) {
    __hip_bfloat16 b = __float2bfloat16(x);
    return *(unsigned short*)&b;
}

// Load an A/B fragment: 2 x b128 from a 16-byte-aligned base.
// A: base = row*HP + half*8 elements; q = base + kb*32;   parts at +0, +16 elem
// B: base = nCol*32 + half*16 elements; q = base + kb*Ncols*32; parts at +0, +8 elem
static __device__ __forceinline__ void load_frag2(const unsigned short* q,
                                                  int elemGap, AFrag& f) {
    *(uint4*)&f.u[0] = *(const uint4*)(q);
    *(uint4*)&f.u[4] = *(const uint4*)(q + elemGap);
}

// ---------------------------------------------------------------------------
// Pack a row-major f32 weight [Kin, NcolsIn] into bf16 WMMA-B fragment layout:
// Bp[(kb*Ncols + n)*32 + (half*16 + e)] = W[kmap(kb,e,half), col(n)]
// headMode=1: column remap n = h*512 + c  ->  W column h*500 + c (for Wg)
// ---------------------------------------------------------------------------
__global__ void pack_b(const float* __restrict__ W, int Kin, int NcolsIn,
                       unsigned short* __restrict__ Bp, int Ncols, int headMode) {
    size_t t = (size_t)blockIdx.x * blockDim.x + threadIdx.x;
    size_t total = (size_t)KBLOCKS * Ncols * 32;
    if (t >= total) return;
    int j    = (int)(t & 31);
    int n    = (int)((t >> 5) % Ncols);
    int kb   = (int)(t / ((size_t)Ncols * 32));
    int e    = j & 15, half = j >> 4;
    int v    = e >> 1, slot = e & 1;
    int k = kb * 32 + ((v & 4) << 2) + ((v & 3) << 1) + slot + (half << 3);
    float val = 0.f;
    int colIn; bool valid;
    if (headMode) {
        int c = n & 511, h = n >> 9;
        valid = (c < H) && (h < HEADS);
        colIn = h * H + c;
    } else {
        valid = (n < NcolsIn);
        colIn = n;
    }
    if (k < Kin && valid) val = W[(size_t)k * NcolsIn + colIn];
    Bp[t] = f2bf(val);
}

__global__ void bias_sum_pad(const float* a, const float* b, float* out,
                             int valid, int total) {
    int t = blockIdx.x * blockDim.x + threadIdx.x;
    if (t >= total) return;
    float v = 0.f;
    if (t < valid) {
        if (a) v += a[t];
        if (b) v += b[t];
    }
    out[t] = v;
}

__global__ void embed_kernel(const int* __restrict__ ids,
                             const float* __restrict__ emb,
                             unsigned short* __restrict__ xb) {
    int t = blockIdx.x * blockDim.x + threadIdx.x; // N*512 exact
    int n = t >> 9, c = t & 511;
    float v = (c < H) ? emb[(size_t)ids[n] * H + c] : 0.f;
    xb[t] = f2bf(v);
}

__global__ void init_state(unsigned short* __restrict__ hb,
                           float* __restrict__ cst) {
    int t = blockIdx.x * blockDim.x + threadIdx.x; // N*512 exact
    hb[t] = 0;       // bf16 +0.0
    cst[t] = 0.f;
}

// ---------------------------------------------------------------------------
// bf16 WMMA GEMM:  C[m,n] = A1[row1(m),:]*B1 (+ A2[m,:]*B2) + bias
// DUAL / GATHER are compile-time: no branches around WMMA, EXEC stays all-1s.
// Software-pipelined over kb: issue kb+1 loads before consuming kb fragments.
// Block = 256 threads = 8 waves; wave tile 16x16; block tile 64(M) x 32(N).
// ---------------------------------------------------------------------------
template <bool DUAL, bool GATHER>
__global__ __launch_bounds__(256)
void gemm_wmma(const unsigned short* __restrict__ A1,
               const int* __restrict__ idx1, int idxOff,
               const unsigned short* __restrict__ A2,
               const unsigned short* __restrict__ B1,
               const unsigned short* __restrict__ B2,
               const float* __restrict__ bias,
               float* __restrict__ Cf, int ldc,
               unsigned short* __restrict__ Cbf, int Ncols, int doRelu) {
    int lane = threadIdx.x & 31;
    int wave = threadIdx.x >> 5;
    int half = lane >> 4, ml = lane & 15;
    int mBase = blockIdx.x * 64 + (wave & 3) * 16;
    int nBase = blockIdx.y * 32 + (wave >> 2) * 16;
    int mRow = mBase + ml;       // this lane's A row
    int nCol = nBase + ml;       // this lane's B column

    int r1 = GATHER ? idx1[(size_t)mRow * DDEG + idxOff] : mRow;
    const unsigned short* a1p = A1 + (size_t)r1 * HP + (half << 3);
    const unsigned short* a2p = DUAL ? (A2 + (size_t)mRow * HP + (half << 3)) : nullptr;
    const unsigned short* b1p = B1 + ((size_t)nCol << 5) + (half << 4);
    const unsigned short* b2p = DUAL ? (B2 + ((size_t)nCol << 5) + (half << 4)) : nullptr;
    const size_t bStep = (size_t)Ncols << 5;   // B elements per kb step

    AFrag a1c, b1c, a2c, b2c;
    load_frag2(a1p, 16, a1c);
    load_frag2(b1p, 8, b1c);
    if (DUAL) {
        load_frag2(a2p, 16, a2c);
        load_frag2(b2p, 8, b2c);
    }

    v8f acc = {};
#pragma unroll
    for (int kb = 0; kb < KBLOCKS; ++kb) {
        AFrag a1n, b1n, a2n, b2n;
        if (kb + 1 < KBLOCKS) {
            load_frag2(a1p + (kb + 1) * 32, 16, a1n);
            load_frag2(b1p + (size_t)(kb + 1) * bStep, 8, b1n);
            if (DUAL) {
                load_frag2(a2p + (kb + 1) * 32, 16, a2n);
                load_frag2(b2p + (size_t)(kb + 1) * bStep, 8, b2n);
            }
        }
        acc = __builtin_amdgcn_wmma_f32_16x16x32_bf16(
            false, a1c.v, false, b1c.v, (short)0, acc, false, false);
        if (DUAL)
            acc = __builtin_amdgcn_wmma_f32_16x16x32_bf16(
                false, a2c.v, false, b2c.v, (short)0, acc, false, false);
        a1c = a1n; b1c = b1n;
        if (DUAL) { a2c = a2n; b2c = b2n; }
    }

    float bval = bias ? bias[nCol] : 0.f;
#pragma unroll
    for (int r = 0; r < 8; ++r) {
        int m = mBase + r + (half << 3);
        float val = acc[r] + bval;
        if (doRelu) val = fmaxf(val, 0.f);
        if (Cf)  Cf[(size_t)m * ldc + nCol] = val;
        if (Cbf) Cbf[(size_t)m * HP + nCol] = f2bf(val);
    }
}

// LSTM cell elementwise (torch gate order i,f,g,o)
__global__ void lstm_cell(const float* __restrict__ gates,
                          float* __restrict__ cst,
                          unsigned short* __restrict__ hb) {
    int t = blockIdx.x * blockDim.x + threadIdx.x;
    if (t >= N_NODES * H) return;
    int n = t / H, c = t % H;
    const float* g = gates + (size_t)n * GATE_PAD;
    float gi = g[c], gf = g[H + c], gg = g[2 * H + c], go = g[3 * H + c];
    float cv = cst[(size_t)n * HP + c];
    float si = 1.f / (1.f + __expf(-gi));
    float sf = 1.f / (1.f + __expf(-gf));
    float so = 1.f / (1.f + __expf(-go));
    cv = sf * cv + si * tanhf(gg);
    float hv = so * tanhf(cv);
    cst[(size_t)n * HP + c] = cv;
    hb[(size_t)n * HP + c] = f2bf(hv);
}

// GAT attention scores: wave per (node, head)
__global__ void gat_scores(const float* __restrict__ z,
                           const float* __restrict__ al,
                           const float* __restrict__ ar,
                           float* __restrict__ el, float* __restrict__ er) {
    int wid = (int)((blockIdx.x * (size_t)blockDim.x + threadIdx.x) >> 5);
    int lane = threadIdx.x & 31;
    if (wid >= N_NODES * HEADS) return;
    int n = wid / HEADS, h = wid % HEADS;
    const float* zr = z + (size_t)n * Z_PAD + h * HP;
    float sl = 0.f, sr = 0.f;
    for (int c = lane; c < H; c += 32) {
        float zv = zr[c];
        sl += zv * al[h * H + c];
        sr += zv * ar[h * H + c];
    }
    for (int o = 16; o > 0; o >>= 1) {
        sl += __shfl_down(sl, o, 32);
        sr += __shfl_down(sr, o, 32);
    }
    if (lane == 0) { el[wid] = sl; er[wid] = sr; }
}

// Edge softmax + alpha-weighted neighbor aggregation + relu + head-mean + elu
__global__ void gat_aggregate(const float* __restrict__ z,
                              const int* __restrict__ nbr,
                              const float* __restrict__ el,
                              const float* __restrict__ er,
                              float* __restrict__ xm) {
    int n = blockIdx.x;
    __shared__ float s_e[HEADS][DDEG];
    __shared__ float s_alpha[HEADS][DDEG];
    __shared__ int s_nb[DDEG];
    int t = threadIdx.x;
    if (t < DDEG) s_nb[t] = nbr[n * DDEG + t];
    __syncthreads();
    if (t < HEADS * DDEG) {
        int h = t / DDEG, d = t % DDEG;
        float ev = el[s_nb[d] * HEADS + h] + er[n * HEADS + h];
        s_e[h][d] = (ev < 0.f) ? SLOPE * ev : ev;   // leaky_relu
    }
    __syncthreads();
    if (t < HEADS) {
        float mx = -1e30f;
        for (int d = 0; d < DDEG; ++d) mx = fmaxf(mx, s_e[t][d]);
        float s = 0.f;
        for (int d = 0; d < DDEG; ++d) {
            float e = __expf(s_e[t][d] - mx);
            s_alpha[t][d] = e; s += e;
        }
        float inv = 1.f / s;
        for (int d = 0; d < DDEG; ++d) s_alpha[t][d] *= inv;
    }
    __syncthreads();
    for (int c = t; c < H; c += blockDim.x) {
        float msum = 0.f;
        for (int h = 0; h < HEADS; ++h) {
            float v = 0.f;
            for (int d = 0; d < DDEG; ++d)
                v += s_alpha[h][d] * z[(size_t)s_nb[d] * Z_PAD + h * HP + c];
            msum += fmaxf(v, 0.f);                   // relu on GAT output
        }
        msum *= (1.f / HEADS);                       // mean over heads
        msum = (msum >= 0.f) ? msum : (__expf(msum) - 1.f);  // elu
        xm[(size_t)n * HP + c] = msum;
    }
}

__global__ void zero_out(float* out, int n) {
    int t = blockIdx.x * blockDim.x + threadIdx.x;
    if (t < n) out[t] = 0.f;
}

// latent = xm . Wr + br, segment-sum by graph id (wave per node)
__global__ void readout(const float* __restrict__ xm,
                        const float* __restrict__ Wr,
                        const float* __restrict__ br,
                        const int* __restrict__ gids,
                        float* __restrict__ out) {
    int wid = (int)((blockIdx.x * (size_t)blockDim.x + threadIdx.x) >> 5);
    int lane = threadIdx.x & 31;
    if (wid >= N_NODES) return;
    float s = 0.f;
    for (int c = lane; c < H; c += 32)
        s += xm[(size_t)wid * HP + c] * Wr[c];
    for (int o = 16; o > 0; o >>= 1) s += __shfl_down(s, o, 32);
    if (lane == 0) atomicAdd(&out[gids[wid]], s + br[0]);
}

extern "C" void kernel_launch(void* const* d_in, const int* in_sizes, int n_in,
                              void* d_out, int out_size, void* d_ws, size_t ws_size,
                              hipStream_t stream) {
    (void)in_sizes; (void)n_in; (void)ws_size;
    const int*   node_ids  = (const int*)d_in[0];
    const int*   neighbors = (const int*)d_in[1];
    const int*   graph_ids = (const int*)d_in[2];
    const float* emb       = (const float*)d_in[4];
    const float *Wih[3], *Whh[3], *bih[3], *bhh[3], *Wself[3], *Wneigh[3], *bl[3];
    for (int l = 0; l < 3; ++l) {
        int b = 5 + 7 * l;
        Wih[l]    = (const float*)d_in[b + 0];
        Whh[l]    = (const float*)d_in[b + 1];
        bih[l]    = (const float*)d_in[b + 2];
        bhh[l]    = (const float*)d_in[b + 3];
        Wself[l]  = (const float*)d_in[b + 4];
        Wneigh[l] = (const float*)d_in[b + 5];
        bl[l]     = (const float*)d_in[b + 6];
    }
    const float* Wg = (const float*)d_in[26];
    const float* al = (const float*)d_in[27];
    const float* ar = (const float*)d_in[28];
    const float* Wr = (const float*)d_in[29];
    const float* br = (const float*)d_in[30];

    char* ws = (char*)d_ws;
    size_t off = 0;
    auto alloc = [&](size_t bytes) {
        size_t o = off;
        off = (off + bytes + 255) & ~(size_t)255;
        return o;
    };

    unsigned short* xbA = (unsigned short*)(ws + alloc((size_t)N_NODES * HP * 2));
    unsigned short* xbB = (unsigned short*)(ws + alloc((size_t)N_NODES * HP * 2));
    unsigned short* hb  = (unsigned short*)(ws + alloc((size_t)N_NODES * HP * 2));
    float* cst = (float*)(ws + alloc((size_t)N_NODES * HP * 4));
    // union buffer: LSTM gates (stride 2048) during layers, GAT z (stride 2560) after
    float* big = (float*)(ws + alloc((size_t)N_NODES * Z_PAD * 4));
    float* xm  = (float*)(ws + alloc((size_t)N_NODES * HP * 4));
    float* elb = (float*)(ws + alloc((size_t)N_NODES * HEADS * 4));
    float* erb = (float*)(ws + alloc((size_t)N_NODES * HEADS * 4));
    unsigned short *WihP[3], *WhhP[3], *WselfP[3], *WneighP[3];
    float *bsum[3], *bpad[3];
    for (int l = 0; l < 3; ++l) {
        WihP[l]    = (unsigned short*)(ws + alloc((size_t)KBLOCKS * GATE_PAD * 32 * 2));
        WhhP[l]    = (unsigned short*)(ws + alloc((size_t)KBLOCKS * GATE_PAD * 32 * 2));
        WselfP[l]  = (unsigned short*)(ws + alloc((size_t)KBLOCKS * HP * 32 * 2));
        WneighP[l] = (unsigned short*)(ws + alloc((size_t)KBLOCKS * HP * 32 * 2));
        bsum[l]    = (float*)(ws + alloc(GATE_PAD * 4));
        bpad[l]    = (float*)(ws + alloc(HP * 4));
    }
    unsigned short* WgP = (unsigned short*)(ws + alloc((size_t)KBLOCKS * Z_PAD * 32 * 2));

    // --- pack weights (L2-resident, reused across all 24 recurrence steps) ---
    for (int l = 0; l < 3; ++l) {
        size_t tg = (size_t)KBLOCKS * GATE_PAD * 32;
        pack_b<<<(unsigned)((tg + 255) / 256), 256, 0, stream>>>(Wih[l], H, GATE_COLS, WihP[l], GATE_PAD, 0);
        pack_b<<<(unsigned)((tg + 255) / 256), 256, 0, stream>>>(Whh[l], H, GATE_COLS, WhhP[l], GATE_PAD, 0);
        size_t ts = (size_t)KBLOCKS * HP * 32;
        pack_b<<<(unsigned)((ts + 255) / 256), 256, 0, stream>>>(Wself[l], H, H, WselfP[l], HP, 0);
        pack_b<<<(unsigned)((ts + 255) / 256), 256, 0, stream>>>(Wneigh[l], H, H, WneighP[l], HP, 0);
        bias_sum_pad<<<(GATE_PAD + 255) / 256, 256, 0, stream>>>(bih[l], bhh[l], bsum[l], GATE_COLS, GATE_PAD);
        bias_sum_pad<<<(HP + 255) / 256, 256, 0, stream>>>(bl[l], nullptr, bpad[l], H, HP);
    }
    {
        size_t tz = (size_t)KBLOCKS * Z_PAD * 32;
        pack_b<<<(unsigned)((tz + 255) / 256), 256, 0, stream>>>(Wg, H, HEADS * H, WgP, Z_PAD, 1);
    }

    // --- embedding lookup -> bf16 activations ---
    embed_kernel<<<(N_NODES * HP) / 256, 256, 0, stream>>>(node_ids, emb, xbA);

    unsigned short* xcur = xbA;
    unsigned short* xnext = xbB;
    dim3 gGates(N_NODES / 64, GATE_PAD / 32);
    dim3 gOut(N_NODES / 64, HP / 32);

    for (int l = 0; l < 3; ++l) {
        init_state<<<(N_NODES * HP) / 256, 256, 0, stream>>>(hb, cst);
        for (int t = 0; t < DDEG; ++t) {
            // gates = x[neighbors[:,t]] @ Wih + h @ Whh + (bih+bhh)
            gemm_wmma<true, true><<<gGates, 256, 0, stream>>>(
                xcur, neighbors, t, hb,
                WihP[l], WhhP[l], bsum[l],
                big, GATE_PAD, nullptr, GATE_PAD, 0);
            lstm_cell<<<((N_NODES * H) + 255) / 256, 256, 0, stream>>>(big, cst, hb);
        }
        // xnext = relu(x @ Wself + hN @ Wneigh + b)  (bf16 out, zero-padded cols)
        gemm_wmma<true, false><<<gOut, 256, 0, stream>>>(
            xcur, nullptr, 0, hb,
            WselfP[l], WneighP[l], bpad[l],
            nullptr, 0, xnext, HP, 1);
        unsigned short* tmp = xcur; xcur = xnext; xnext = tmp;
    }

    // --- GAT: z = x @ Wg (head-major, per-head stride 512) ---
    dim3 gZ(N_NODES / 64, Z_PAD / 32);
    gemm_wmma<false, false><<<gZ, 256, 0, stream>>>(
        xcur, nullptr, 0, nullptr,
        WgP, nullptr, nullptr,
        big, Z_PAD, nullptr, Z_PAD, 0);
    gat_scores<<<(N_NODES * HEADS * 32) / 256, 256, 0, stream>>>(big, al, ar, elb, erb);
    gat_aggregate<<<N_NODES, 256, 0, stream>>>(big, neighbors, elb, erb, xm);

    // --- readout + segment sum ---
    zero_out<<<1, 64, 0, stream>>>((float*)d_out, out_size);
    readout<<<(N_NODES * 32) / 256, 256, 0, stream>>>(xm, Wr, br, graph_ids, (float*)d_out);
}
